// TransformerBlock_4810363372375
// MI455X (gfx1250) — matrix operand order repaired and checked by reading, not compiled, not run on hardware
//
#include <hip/hip_runtime.h>
#include <hip/hip_bf16.h>

typedef __bf16 bf16;
typedef __attribute__((ext_vector_type(16))) __bf16 v16bf;
typedef __attribute__((ext_vector_type(8)))  float  v8f;

#define DIMD   768
#define SEQ    2048
#define BATCH  2
#define NT     (BATCH * SEQ)   /* 4096 tokens */
#define NHEADS 12
#define HD     64
#define FF     3072
#define LN_EPS 1e-3f
#define NEGMASK (-1e9f)

union FragU { v16bf v; uint4 q[2]; };

// ---------------------------------------------------------------------------
// Weight convert + transpose: in[K][N] fp32 -> out[N][K] bf16
// ---------------------------------------------------------------------------
__global__ void __launch_bounds__(256) convt_kernel(const float* __restrict__ in,
                                                    bf16* __restrict__ out,
                                                    int K, int N) {
  int i = blockIdx.x * 256 + threadIdx.x;
  if (i >= K * N) return;
  int k = i / N, n = i - k * N;
  out[(size_t)n * K + k] = (bf16)in[i];
}

// ---------------------------------------------------------------------------
// LayerNorm: x[row][768] fp32 -> out bf16 (gamma/beta applied, eps=1e-3)
// one block (256 thr) per row, 3 elements/thread
// ---------------------------------------------------------------------------
__global__ void __launch_bounds__(256) ln_kernel(const float* __restrict__ x,
                                                 const float* __restrict__ g,
                                                 const float* __restrict__ b,
                                                 bf16* __restrict__ out) {
  __shared__ float s_sum[8], s_sq[8];
  const int row = blockIdx.x;
  const float* xr = x + (size_t)row * DIMD;
  float v[3], sum = 0.f, sq = 0.f;
#pragma unroll
  for (int i = 0; i < 3; ++i) {
    v[i] = xr[threadIdx.x + 256 * i];
    sum += v[i]; sq += v[i] * v[i];
  }
#pragma unroll
  for (int m = 1; m < 32; m <<= 1) {
    sum += __shfl_xor(sum, m, 32);
    sq  += __shfl_xor(sq,  m, 32);
  }
  if ((threadIdx.x & 31) == 0) { s_sum[threadIdx.x >> 5] = sum; s_sq[threadIdx.x >> 5] = sq; }
  __syncthreads();
  sum = 0.f; sq = 0.f;
#pragma unroll
  for (int i = 0; i < 8; ++i) { sum += s_sum[i]; sq += s_sq[i]; }
  const float mu   = sum * (1.0f / DIMD);
  const float var  = sq * (1.0f / DIMD) - mu * mu;
  const float rstd = rsqrtf(var + LN_EPS);
#pragma unroll
  for (int i = 0; i < 3; ++i) {
    int c = threadIdx.x + 256 * i;
    out[(size_t)row * DIMD + c] = (bf16)((v[i] - mu) * rstd * g[c] + b[c]);
  }
}

// ---------------------------------------------------------------------------
// bf16 WMMA GEMM: C[M,N] = A[M,K] * Bt[N,K]^T (+bias)(+relu)(+resid)
// one wave computes a 32x64 tile (2 A-frags x 4 B-frags = 8 WMMA / k-step);
// 8 waves / block.
// FLAGS: 1=BIAS 2=RELU 4=RESID(fp32 [M,N]) 8=OUT_BF16
//        16=OUT_VT([B,H,d,S] bf16) 32=SCALE_OUT(*1/8, for Q)
// ---------------------------------------------------------------------------
template <int FLAGS>
__global__ void __launch_bounds__(256) gemm_bf16_kernel(
    const bf16* __restrict__ A, const bf16* __restrict__ Bt,
    const float* __restrict__ bias, const float* __restrict__ resid,
    void* __restrict__ out, int M, int N, int K) {
  constexpr bool BIAS   = (FLAGS & 1) != 0;
  constexpr bool RELU   = (FLAGS & 2) != 0;
  constexpr bool RESID  = (FLAGS & 4) != 0;
  constexpr bool OUTBF  = (FLAGS & 8) != 0;
  constexpr bool VTOUT  = (FLAGS & 16) != 0;
  constexpr bool SCALEO = (FLAGS & 32) != 0;

  const int lane   = threadIdx.x & 31;
  const int w      = blockIdx.x * 8 + (threadIdx.x >> 5);
  const int tilesN = N >> 6;
  if (w >= (M >> 5) * tilesN) return;
  const int tm   = w / tilesN;               // 32-row tile index
  const int n0   = (w - tm * tilesN) << 6;
  const int half = lane >> 4;
  const int l15  = lane & 15;

  v8f acc[2][4] = {};
  const bf16* arow0 = A + (size_t)((tm << 5) + l15) * K;
  const bf16* arow1 = arow0 + (size_t)16 * K;
  for (int k0 = 0; k0 < K; k0 += 32) {
    FragU fa0, fa1;
    fa0.q[0] = *(const uint4*)(arow0 + k0 + 8 * half);
    fa0.q[1] = *(const uint4*)(arow0 + k0 + 16 + 8 * half);
    fa1.q[0] = *(const uint4*)(arow1 + k0 + 8 * half);
    fa1.q[1] = *(const uint4*)(arow1 + k0 + 16 + 8 * half);
#pragma unroll
    for (int j = 0; j < 4; ++j) {
      const bf16* bp = Bt + (size_t)(n0 + (j << 4) + l15) * K + k0 + 8 * half;
      FragU fb;
      fb.q[0] = *(const uint4*)bp;
      fb.q[1] = *(const uint4*)(bp + 16);
      acc[0][j] = __builtin_amdgcn_wmma_f32_16x16x32_bf16(
          false, fa0.v, false, fb.v, (short)0, acc[0][j], false, false);
      acc[1][j] = __builtin_amdgcn_wmma_f32_16x16x32_bf16(
          false, fa1.v, false, fb.v, (short)0, acc[1][j], false, false);
    }
  }
#pragma unroll
  for (int i = 0; i < 2; ++i) {
#pragma unroll
    for (int j = 0; j < 4; ++j) {
#pragma unroll
      for (int r = 0; r < 8; ++r) {
        const int row = (tm << 5) + (i << 4) + r + 8 * half;
        const int col = n0 + (j << 4) + l15;
        float v = acc[i][j][r];
        if (SCALEO) v *= 0.125f;
        if (BIAS)   v += bias[col];
        if (RELU)   v = fmaxf(v, 0.0f);
        if (RESID)  v += resid[(size_t)row * N + col];
        if (VTOUT) {
          const int bb = row >> 11, s = row & (SEQ - 1);
          const int hh = col >> 6,  dc = col & 63;
          ((bf16*)out)[((size_t)((bb * NHEADS + hh) * HD + dc)) * SEQ + s] = (bf16)v;
        } else if (OUTBF) {
          ((bf16*)out)[(size_t)row * N + col] = (bf16)v;
        } else {
          ((float*)out)[(size_t)row * N + col] = v;
        }
      }
    }
  }
}

// ---------------------------------------------------------------------------
// Causal flash attention, one wave per 16-query tile of one (b,h).
// Q (pre-scaled by 1/8), K: [NT, 768] bf16 token-major.
// Vt: [B,H,HD,SEQ] bf16.  O: [NT,768] bf16.
// All K and V fragments for a 32-key step are issued up front so their
// latency overlaps the score WMMAs and the softmax VALU/TRANS work.
// ---------------------------------------------------------------------------
__global__ void __launch_bounds__(128) attn_kernel(const bf16* __restrict__ Q,
                                                   const bf16* __restrict__ Kc,
                                                   const bf16* __restrict__ Vt,
                                                   bf16* __restrict__ O) {
  __shared__ __align__(16) bf16 plds[4][16][32];
  const int lane = threadIdx.x & 31;
  const int wloc = threadIdx.x >> 5;
  const int w    = blockIdx.x * 4 + wloc;
  const int S16  = SEQ / 16;
  const int b    = w / (NHEADS * S16);
  const int rem  = w - b * (NHEADS * S16);
  const int h    = rem / S16;
  const int q0   = (rem - h * S16) << 4;
  const int half = lane >> 4;
  const int l15  = lane & 15;

  // Q fragments for this 16-row tile (K-dim = 64 -> 2 fragments)
  const bf16* qrow = Q + (size_t)(b * SEQ + q0 + l15) * DIMD + h * HD;
  v16bf qa[2];
#pragma unroll
  for (int s = 0; s < 2; ++s) {
    FragU f;
    f.q[0] = *(const uint4*)(qrow + 32 * s + 8 * half);
    f.q[1] = *(const uint4*)(qrow + 32 * s + 16 + 8 * half);
    qa[s] = f.v;
  }

  v8f oacc[4] = {};
  float mrow[8], lrow[8];
#pragma unroll
  for (int r = 0; r < 8; ++r) { mrow[r] = -1e30f; lrow[r] = 0.f; }

  const bf16* kbase = Kc + (size_t)(b * SEQ) * DIMD + h * HD;
  const bf16* vbase = Vt + (size_t)((b * NHEADS + h) * HD) * SEQ;

  for (int kt = 0; kt <= q0 + 15; kt += 32) {
    // ---- issue ALL loads for this key-step first: 4 K frags + 4 V frags
    FragU fk[2][2];   // [key-tile][d-chunk]
#pragma unroll
    for (int s = 0; s < 2; ++s) {
      const bf16* kp0 = kbase + (size_t)(kt + l15) * DIMD + 32 * s + 8 * half;
      const bf16* kp1 = kbase + (size_t)(kt + 16 + l15) * DIMD + 32 * s + 8 * half;
      fk[0][s].q[0] = *(const uint4*)kp0;  fk[0][s].q[1] = *(const uint4*)(kp0 + 16);
      fk[1][s].q[0] = *(const uint4*)kp1;  fk[1][s].q[1] = *(const uint4*)(kp1 + 16);
    }
    FragU fv[4];      // V fragments, consumed only after softmax
#pragma unroll
    for (int j = 0; j < 4; ++j) {
      const bf16* vp = vbase + (size_t)((j << 4) + l15) * SEQ + kt + 8 * half;
      fv[j].q[0] = *(const uint4*)vp;
      fv[j].q[1] = *(const uint4*)(vp + 16);
    }
    // ---- scores: two 16x16 tiles (Q pre-scaled by 1/sqrt(64))
    v8f s0 = {}, s1 = {};
#pragma unroll
    for (int s = 0; s < 2; ++s) {
      s0 = __builtin_amdgcn_wmma_f32_16x16x32_bf16(false, qa[s], false, fk[0][s].v, (short)0, s0, false, false);
      s1 = __builtin_amdgcn_wmma_f32_16x16x32_bf16(false, qa[s], false, fk[1][s].v, (short)0, s1, false, false);
    }
    // ---- causal mask
    float s0v[8], s1v[8];
#pragma unroll
    for (int r = 0; r < 8; ++r) { s0v[r] = s0[r]; s1v[r] = s1[r]; }
    if (kt + 31 > q0) {
      const int k0c = kt + l15;
#pragma unroll
      for (int r = 0; r < 8; ++r) {
        const int qr = q0 + r + 8 * half;
        if (k0c > qr)      s0v[r] = NEGMASK;
        if (k0c + 16 > qr) s1v[r] = NEGMASK;
      }
    }
    // ---- online softmax (row stats across 16-lane halves)
    float p0[8], p1[8], fs[8];
#pragma unroll
    for (int r = 0; r < 8; ++r) {
      float mx = fmaxf(s0v[r], s1v[r]);
#pragma unroll
      for (int m = 1; m < 16; m <<= 1) mx = fmaxf(mx, __shfl_xor(mx, m, 32));
      const float mnew = fmaxf(mrow[r], mx);
      const float f  = __expf(mrow[r] - mnew);
      const float e0 = __expf(s0v[r] - mnew);
      const float e1 = __expf(s1v[r] - mnew);
      float rs = e0 + e1;
#pragma unroll
      for (int m = 1; m < 16; m <<= 1) rs += __shfl_xor(rs, m, 32);
      lrow[r] = lrow[r] * f + rs;
      mrow[r] = mnew; fs[r] = f; p0[r] = e0; p1[r] = e1;
    }
#pragma unroll
    for (int j = 0; j < 4; ++j)
#pragma unroll
      for (int r = 0; r < 8; ++r) oacc[j][r] *= fs[r];
    // ---- P: C-layout -> A-layout via per-wave LDS staging (16x32 bf16)
#pragma unroll
    for (int r = 0; r < 8; ++r) {
      plds[wloc][r + 8 * half][l15]      = (bf16)p0[r];
      plds[wloc][r + 8 * half][16 + l15] = (bf16)p1[r];
    }
    asm volatile("s_wait_dscnt 0" ::: "memory");
    FragU pf;
    pf.q[0] = *(const uint4*)&plds[wloc][l15][8 * half];
    pf.q[1] = *(const uint4*)&plds[wloc][l15][16 + 8 * half];
    // ---- O += P(16x32) x V(32x64)  (V frags already resident)
#pragma unroll
    for (int j = 0; j < 4; ++j)
      oacc[j] = __builtin_amdgcn_wmma_f32_16x16x32_bf16(false, pf.v, false, fv[j].v, (short)0, oacc[j], false, false);
  }
  // ---- normalize and store [token, h*64 + dcol]
  float linv[8];
#pragma unroll
  for (int r = 0; r < 8; ++r) linv[r] = 1.0f / lrow[r];
  bf16* obase = O + (size_t)(b * SEQ + q0) * DIMD + h * HD;
#pragma unroll
  for (int r = 0; r < 8; ++r) {
    bf16* orow = obase + (size_t)(r + 8 * half) * DIMD;
#pragma unroll
    for (int j = 0; j < 4; ++j) orow[(j << 4) + l15] = (bf16)(oacc[j][r] * linv[r]);
  }
}

// ---------------------------------------------------------------------------
extern "C" void kernel_launch(void* const* d_in, const int* in_sizes, int n_in,
                              void* d_out, int out_size, void* d_ws, size_t ws_size,
                              hipStream_t stream) {
  const float* x    = (const float*)d_in[0];
  const float* W_Q  = (const float*)d_in[1];
  const float* W_K  = (const float*)d_in[2];
  const float* W_V  = (const float*)d_in[3];
  const float* W_O  = (const float*)d_in[4];
  const float* w1   = (const float*)d_in[5];
  const float* b1   = (const float*)d_in[6];
  const float* w2   = (const float*)d_in[7];
  const float* b2   = (const float*)d_in[8];
  const float* ln1g = (const float*)d_in[9];
  const float* ln1b = (const float*)d_in[10];
  const float* ln2g = (const float*)d_in[11];
  const float* ln2b = (const float*)d_in[12];
  (void)in_sizes; (void)n_in; (void)out_size; (void)ws_size;

  char* p = (char*)d_ws;
  auto alloc = [&](size_t bytes) { void* r = (void*)p; p += (bytes + 255) & ~(size_t)255; return r; };
  bf16*  WtQ    = (bf16*)alloc((size_t)DIMD * DIMD * 2);
  bf16*  WtK    = (bf16*)alloc((size_t)DIMD * DIMD * 2);
  bf16*  WtV    = (bf16*)alloc((size_t)DIMD * DIMD * 2);
  bf16*  WtO    = (bf16*)alloc((size_t)DIMD * DIMD * 2);
  bf16*  wt1    = (bf16*)alloc((size_t)DIMD * FF * 2);
  bf16*  wt2    = (bf16*)alloc((size_t)FF * DIMD * 2);
  bf16*  hbuf   = (bf16*)alloc((size_t)NT * DIMD * 2);
  bf16*  Qb     = (bf16*)alloc((size_t)NT * DIMD * 2);
  bf16*  Kb     = (bf16*)alloc((size_t)NT * DIMD * 2);
  bf16*  Vt     = (bf16*)alloc((size_t)NT * DIMD * 2);
  bf16*  valsb  = (bf16*)alloc((size_t)NT * DIMD * 2);
  float* x1     = (float*)alloc((size_t)NT * DIMD * 4);
  bf16*  h2buf  = (bf16*)alloc((size_t)NT * DIMD * 2);
  bf16*  hidden = (bf16*)alloc((size_t)NT * FF * 2);

  // weights -> bf16 transposed [N,K]
  convt_kernel<<<(DIMD * DIMD + 255) / 256, 256, 0, stream>>>(W_Q, WtQ, DIMD, DIMD);
  convt_kernel<<<(DIMD * DIMD + 255) / 256, 256, 0, stream>>>(W_K, WtK, DIMD, DIMD);
  convt_kernel<<<(DIMD * DIMD + 255) / 256, 256, 0, stream>>>(W_V, WtV, DIMD, DIMD);
  convt_kernel<<<(DIMD * DIMD + 255) / 256, 256, 0, stream>>>(W_O, WtO, DIMD, DIMD);
  convt_kernel<<<(DIMD * FF   + 255) / 256, 256, 0, stream>>>(w1,  wt1,  DIMD, FF);
  convt_kernel<<<(FF   * DIMD + 255) / 256, 256, 0, stream>>>(w2,  wt2,  FF,   DIMD);

  // LN1
  ln_kernel<<<NT, 256, 0, stream>>>(x, ln1g, ln1b, hbuf);

  // Q/K/V projections (Q pre-scaled by 1/8; V written in [B,H,d,S] layout)
  const int gQ = ((NT / 32) * (DIMD / 64)) / 8;   // 192 blocks
  gemm_bf16_kernel<8 | 32><<<gQ, 256, 0, stream>>>(hbuf, WtQ, nullptr, nullptr, Qb, NT, DIMD, DIMD);
  gemm_bf16_kernel<8     ><<<gQ, 256, 0, stream>>>(hbuf, WtK, nullptr, nullptr, Kb, NT, DIMD, DIMD);
  gemm_bf16_kernel<16    ><<<gQ, 256, 0, stream>>>(hbuf, WtV, nullptr, nullptr, Vt, NT, DIMD, DIMD);

  // causal flash attention
  attn_kernel<<<(NT * NHEADS / 16) / 4, 128, 0, stream>>>(Qb, Kb, Vt, valsb);

  // O projection + residual (fp32 x1)
  gemm_bf16_kernel<4><<<gQ, 256, 0, stream>>>(valsb, WtO, nullptr, x, x1, NT, DIMD, DIMD);

  // LN2
  ln_kernel<<<NT, 256, 0, stream>>>(x1, ln2g, ln2b, h2buf);

  // FFN
  const int gF = ((NT / 32) * (FF / 64)) / 8;     // 768 blocks
  gemm_bf16_kernel<1 | 2 | 8><<<gF, 256, 0, stream>>>(h2buf, wt1, b1, nullptr, hidden, NT, FF, DIMD);
  gemm_bf16_kernel<1 | 4    ><<<gQ, 256, 0, stream>>>(hidden, wt2, b2, x1, d_out, NT, DIMD, FF);
}
